// Resnet50_MLP_20993800143032
// MI455X (gfx1250) — matrix operand order repaired and checked by reading, not compiled, not run on hardware
//
#include <hip/hip_runtime.h>
#include <hip/hip_bf16.h>
#include <math.h>

// ============================================================================
// ResNet50 -> LSTM -> attention head for MI455X (gfx1250, wave32, WMMA).
// All convs are implicit-GEMM in bf16 via v_wmma_f32_16x16x32_bf16, NHWC
// activations, BN folded into weights/bias on device, residual+ReLU fused
// into the GEMM epilogue. Weight tiles are staged into LDS with the CDNA5
// Tensor Data Mover (tensor_load_to_lds + s_wait_tensorcnt) when available.
// ============================================================================

typedef __attribute__((ext_vector_type(16))) __bf16    v16bf;
typedef __attribute__((ext_vector_type(8)))  float     v8f;
typedef __attribute__((ext_vector_type(8)))  int       v8i;
typedef __attribute__((ext_vector_type(4)))  int       v4i;
typedef __attribute__((ext_vector_type(4)))  unsigned  v4u;
typedef unsigned short bf16_t;   // raw bf16 bits in buffers

#if defined(__has_builtin)
#if __has_builtin(__builtin_amdgcn_tensor_load_to_lds)
#define USE_TDM 1
#else
#define USE_TDM 0
#endif
#else
#define USE_TDM 0
#endif

// ---------------- small helpers ----------------
__device__ __forceinline__ bf16_t f2bf(float f) {
  union { float f; unsigned u; } x; x.f = f;
  unsigned r = x.u + 0x7FFFu + ((x.u >> 16) & 1u);   // round-to-nearest-even
  return (bf16_t)(r >> 16);
}
__device__ __forceinline__ float bf2f(bf16_t h) {
  union { unsigned u; float f; } x; x.u = ((unsigned)h) << 16; return x.f;
}
__device__ __forceinline__ float sigf(float x) { return 1.f / (1.f + expf(-x)); }

__device__ __forceinline__ v8i mk8(int4 a, int4 b) {
  v8i v; v[0]=a.x; v[1]=a.y; v[2]=a.z; v[3]=a.w; v[4]=b.x; v[5]=b.y; v[6]=b.z; v[7]=b.w;
  return v;
}
__device__ __forceinline__ v8f wmma_bf16(v8i a, v8i b, v8f c) {
  return __builtin_amdgcn_wmma_f32_16x16x32_bf16(
      /*neg_a=*/false, __builtin_bit_cast(v16bf, a),
      /*neg_b=*/false, __builtin_bit_cast(v16bf, b),
      /*c_mod=*/(short)0, c, /*reuse_a=*/false, /*reuse_b=*/false);
}

// ============================================================================
// Generic implicit-GEMM conv, bf16 WMMA, NHWC.
//   out[m, co] = sum_k A[m,k] * Wp[k,co] + bias[co] (+res) (+relu)
//   m = ((b*Ho)+ho)*Wo+wo ; k = ((kh*KW)+kw)*Cin + ci  (Cin % 32 == 0)
// Block tile: M=128, N=64.  8 waves, each 32x32 (2x2 WMMA 16x16x32 frags).
// ============================================================================
#define TM 128
#define TN 64
#define AROWB 80      // LDS bytes per A row (32 bf16 padded -> 16B aligned, bank-spread)
#define BROWB 144     // LDS bytes per B row (64 bf16 = 128B + 16B pad)
#define A_BYTES (128 * AROWB)   // 10240
#define B_BYTES (32 * BROWB)    //  4608

struct ConvP {
  const bf16_t* in;     // NHWC bf16
  const bf16_t* wp;     // [K][Cout] bf16 (BN-folded)
  const float*  bias;   // [Cout]
  const bf16_t* res;    // NHWC bf16 residual or nullptr
  bf16_t* out_bf;       // bf16 NHWC output (if !out_f32)
  float*  out_f;        // f32 output (if out_f32)
  int Hin, Win, Cin, Ho, Wo, Cout;
  int KH, KW, stride, pad;
  int M, steps, relu, out_f32;
};

#if USE_TDM
// TDM D# (ISA 8.3/8.4): 2D bf16 tile [tile_dim1=32 rows][tile_dim0=64 cols],
// row stride = Cout elements, LDS rows padded 128B -> 144B via pad fields
// (pad_interval code 4 == 32 dwords, pad_amount code 3 == 4 dwords).
__device__ __forceinline__ void tdm_load_b_tile(unsigned lds_off, const bf16_t* gptr, int cout) {
  const unsigned long long ga = (unsigned long long)(size_t)gptr;
  v4u g0;
  g0[0] = 1u;                                              // count=1 (valid), user D#
  g0[1] = lds_off;                                         // lds_addr (bytes)
  g0[2] = (unsigned)(ga & 0xFFFFFFFFu);                    // global_addr[31:0]
  g0[3] = (unsigned)((ga >> 32) & 0x1FFFFFFu) | (2u << 30);// addr[56:32] | type=2
  v8i g1;
  g1[0] = (1 << 16) | (1 << 20) | (4 << 22) | (3 << 25);   // data_size=2B, pad_en, intv, amt
  const unsigned td0 = 64, td1 = 32;                       // tensor dims (tile always in-bounds)
  g1[1] = (int)((td0 & 0xFFFFu) << 16);                    // dim0[15:0] (barrier addr = 0)
  g1[2] = (int)(((td0 >> 16) & 0xFFFFu) | ((td1 & 0xFFFFu) << 16));
  g1[3] = (int)(((td1 >> 16) & 0xFFFFu) | (64u << 16));    // tile_dim0 = 64
  g1[4] = 32;                                              // tile_dim1 = 32, tile_dim2 = 0
  g1[5] = cout;                                            // tensor_dim0_stride (elems)
  g1[6] = 0;
  g1[7] = 0;
  v4i g2 = {0, 0, 0, 0}, g3 = {0, 0, 0, 0};
#if defined(__clang_major__) && (__clang_major__ >= 23)
  v8i g4 = {0, 0, 0, 0, 0, 0, 0, 0};
  __builtin_amdgcn_tensor_load_to_lds(g0, g1, g2, g3, g4, 0);
#else
  __builtin_amdgcn_tensor_load_to_lds(g0, g1, g2, g3, 0);
#endif
}
#endif

__global__ __launch_bounds__(256) void k_conv_wmma(ConvP p) {
  __shared__ __align__(16) unsigned char sm[2 * (A_BYTES + B_BYTES)];
  unsigned char* const As0 = sm;
  unsigned char* const As1 = sm + A_BYTES;
  unsigned char* const Bs0 = sm + 2 * A_BYTES;
  unsigned char* const Bs1 = sm + 2 * A_BYTES + B_BYTES;

  const int tid    = threadIdx.x;
  const int blockM = blockIdx.y * TM;
  const int blockN = blockIdx.x * TN;

  // ---- A loader: 2 threads per tile row, 16 bf16 each ----
  const int arow = tid >> 1;
  const int aseg = (tid & 1) * 16;          // bf16 offset within 32-elem row
  const int mrow = blockM + arow;
  const bool mval = (mrow < p.M);
  int ab = 0, aho = 0, awo = 0;
  if (mval) {
    const int HoWo = p.Ho * p.Wo;
    ab = mrow / HoWo; int r = mrow - ab * HoWo;
    aho = r / p.Wo;  awo = r - aho * p.Wo;
  }
  const int hbase = aho * p.stride - p.pad;
  const int wbase = awo * p.stride - p.pad;

#if !USE_TDM
  // manual B loader: 8 threads per k row, 8 bf16 each
  const int brow = tid >> 3;
  const int bcol = (tid & 7) * 8;
  int4 brg;
#endif

  // filter-position tracker for the A-load pipeline (ci fastest, then kw, kh)
  int kh = 0, kw = 0, ci0 = 0;
  int4 arg0, arg1;

  auto load_step = [&]() {
    const int hi = hbase + kh, wi = wbase + kw;
    const bool ok = mval && (hi >= 0) && (hi < p.Hin) && (wi >= 0) && (wi < p.Win);
    if (ok) {
      const bf16_t* ap = p.in + ((size_t)(ab * p.Hin + hi) * p.Win + wi) * p.Cin + ci0 + aseg;
      arg0 = *(const int4*)ap;
      arg1 = *(const int4*)(ap + 8);
    } else {
      arg0 = make_int4(0, 0, 0, 0); arg1 = arg0;
    }
#if !USE_TDM
    const size_t krow = (size_t)((kh * p.KW + kw) * p.Cin + ci0) + brow;
    const bf16_t* bp = p.wp + krow * p.Cout + blockN + bcol;
    brg = *(const int4*)bp;
    __builtin_prefetch(bp + (size_t)8 * 32 * p.Cout, 0, 1);
#endif
    ci0 += 32;
    if (ci0 == p.Cin) { ci0 = 0; if (++kw == p.KW) { kw = 0; ++kh; } }
  };
  auto store_step = [&](unsigned char* A, unsigned char* B) {
    unsigned char* ad = A + arow * AROWB + aseg * 2;
    *(int4*)ad = arg0; *(int4*)(ad + 16) = arg1;
#if !USE_TDM
    *(int4*)(B + brow * BROWB + bcol * 2) = brg;
#else
    (void)B;
#endif
  };

#if USE_TDM
  const bool tdmw = (tid < 32);  // wave 0 drives the Tensor Data Mover
  const unsigned bs_off[2] = { (unsigned)(size_t)(void*)Bs0, (unsigned)(size_t)(void*)Bs1 };
  auto issue_b = [&](int sidx, int buf) {
    tdm_load_b_tile(bs_off[buf], p.wp + (size_t)sidx * 32 * p.Cout + blockN, p.Cout);
  };
#endif

  // ---- per-wave compute mapping ----
  const int lane = tid & 31;
  const int l15  = lane & 15;
  const int lhi  = lane >> 4;       // 0/1 fragment halves per ISA 7.12.2
  const int wave = tid >> 5;
  const int wr   = wave & 3;        // 4 M sub-tiles of 32
  const int wc   = wave >> 2;       // 2 N sub-tiles of 32
  v8f acc00 = {}, acc01 = {}, acc10 = {}, acc11 = {};

  auto compute = [&](const unsigned char* A, const unsigned char* B) {
    const unsigned char* a0p = A + (wr * 32 + l15) * AROWB + lhi * 16;
    const unsigned char* a1p = a0p + 16 * AROWB;
    v8i a0 = mk8(*(const int4*)a0p, *(const int4*)(a0p + 32));
    v8i a1 = mk8(*(const int4*)a1p, *(const int4*)(a1p + 32));
    const unsigned char* bp = B + (lhi * 8) * BROWB + (wc * 32 + l15) * 2;
    v8i b0, b1;
    for (int j = 0; j < 8; ++j) {
      const unsigned char* q0 = bp + ((j & 3) * 2 + (j >> 2) * 16) * BROWB;
      b0[j] = (int)((unsigned)*(const unsigned short*)q0 | ((unsigned)*(const unsigned short*)(q0 + BROWB) << 16));
      b1[j] = (int)((unsigned)*(const unsigned short*)(q0 + 32) | ((unsigned)*(const unsigned short*)(q0 + BROWB + 32) << 16));
    }
    acc00 = wmma_bf16(a0, b0, acc00);
    acc01 = wmma_bf16(a0, b1, acc01);
    acc10 = wmma_bf16(a1, b0, acc10);
    acc11 = wmma_bf16(a1, b1, acc11);
  };

  // ---- double-buffered main loop ----
#if USE_TDM
  if (tdmw) issue_b(0, 0);
#endif
  load_step();
  store_step(As0, Bs0);
  for (int s = 0; s < p.steps; ++s) {
#if USE_TDM
    if (tdmw) __builtin_amdgcn_s_wait_tensorcnt(0);   // B[cur] landed in LDS
#endif
    __syncthreads();
    const bool nx = (s + 1 < p.steps);
#if USE_TDM
    if (tdmw && nx) issue_b(s + 1, (s + 1) & 1);      // overlap TDM with WMMAs
#endif
    if (nx) load_step();
    if (s & 1) compute(As1, Bs1); else compute(As0, Bs0);
    if (nx) { if (s & 1) store_step(As0, Bs0); else store_step(As1, Bs1); }
  }

  // ---- epilogue: bias (+res) (+relu), bf16 or f32 store ----
  for (int mi = 0; mi < 2; ++mi)
    for (int nj = 0; nj < 2; ++nj) {
      const v8f& a = (mi == 0) ? (nj == 0 ? acc00 : acc01) : (nj == 0 ? acc10 : acc11);
      const int n = blockN + wc * 32 + nj * 16 + l15;
      const float bsv = p.bias[n];
      const int mbase = blockM + wr * 32 + mi * 16 + lhi * 8;
      for (int r = 0; r < 8; ++r) {
        const int mm = mbase + r;
        if (mm < p.M) {
          const size_t idx = (size_t)mm * p.Cout + n;
          float v = a[r] + bsv;
          if (p.res)  v += bf2f(p.res[idx]);
          if (p.relu) v = fmaxf(v, 0.f);
          if (p.out_f32) p.out_f[idx] = v; else p.out_bf[idx] = f2bf(v);
        }
      }
    }
}

// ============================================================================
// Weight prep: fold BN into conv weights, transpose OIHW f32 -> [K][Co] bf16.
// ============================================================================
__global__ void k_fold_conv(const float* __restrict__ w, const float* __restrict__ g,
                            const float* __restrict__ b, const float* __restrict__ m,
                            const float* __restrict__ v, bf16_t* __restrict__ wp,
                            float* __restrict__ bias, int Co, int Ci, int KH, int KW) {
  const long t = (long)blockIdx.x * 256 + threadIdx.x;
  if (t < (long)Co) {
    const float inv = g[t] / sqrtf(v[t] + 1e-5f);
    bias[t] = b[t] - m[t] * inv;
  }
  const long tot = (long)Co * Ci * KH * KW;
  if (t >= tot) return;
  const int cikk = Ci * KH * KW;
  const int co = (int)(t / cikk); const int r = (int)(t - (long)co * cikk);
  const int ci = r / (KH * KW);   const int rr = r % (KH * KW);
  const int khh = rr / KW, kww = rr % KW;
  const float inv = g[co] / sqrtf(v[co] + 1e-5f);
  const long k = (long)(khh * KW + kww) * Ci + ci;
  wp[k * Co + co] = f2bf(w[t] * inv);
}

// stem: keep OIHW f32, just fold BN scale
__global__ void k_fold_conv1(const float* __restrict__ w, const float* __restrict__ g,
                             const float* __restrict__ b, const float* __restrict__ m,
                             const float* __restrict__ v, float* __restrict__ wf,
                             float* __restrict__ bias) {
  const int t = blockIdx.x * 256 + threadIdx.x;
  if (t < 64) { float inv = g[t] / sqrtf(v[t] + 1e-5f); bias[t] = b[t] - m[t] * inv; }
  if (t >= 64 * 147) return;
  const int co = t / 147;
  const float inv = g[co] / sqrtf(v[co] + 1e-5f);
  wf[t] = w[t] * inv;
}

// ============================================================================
// Stem: conv 7x7/2 pad3 (3 in-ch, direct f32) + BN + ReLU -> NHWC bf16
// ============================================================================
__global__ __launch_bounds__(256) void k_conv1(const float* __restrict__ in,
                                               const float* __restrict__ wf,
                                               const float* __restrict__ bias,
                                               bf16_t* __restrict__ out) {
  const long t = (long)blockIdx.x * 256 + threadIdx.x;   // [M=64*112*112][64]
  const int co = (int)(t & 63);
  const long m = t >> 6;
  const int b = (int)(m / (112 * 112)); const int r = (int)(m % (112 * 112));
  const int ho = r / 112, wo = r % 112;
  float acc = bias[co];
  for (int c = 0; c < 3; ++c)
    for (int khh = 0; khh < 7; ++khh) {
      const int hi = ho * 2 + khh - 3;
      if (hi < 0 || hi >= 224) continue;
      for (int kww = 0; kww < 7; ++kww) {
        const int wi = wo * 2 + kww - 3;
        if (wi < 0 || wi >= 224) continue;
        acc += in[((size_t)(b * 3 + c) * 224 + hi) * 224 + wi] *
               wf[((co * 3 + c) * 7 + khh) * 7 + kww];
      }
    }
  out[m * 64 + co] = f2bf(fmaxf(acc, 0.f));
}

// maxpool 3x3/2 pad1: [64,112,112,64] -> [64,56,56,64], NHWC bf16
__global__ __launch_bounds__(256) void k_pool(const bf16_t* __restrict__ in,
                                              bf16_t* __restrict__ out) {
  const long t = (long)blockIdx.x * 256 + threadIdx.x;
  const int c = (int)(t & 63);
  const long m = t >> 6;
  const int b = (int)(m / (56 * 56)); const int r = (int)(m % (56 * 56));
  const int ho = r / 56, wo = r % 56;
  float best = -3.4e38f;
  for (int khh = 0; khh < 3; ++khh) {
    const int hi = ho * 2 + khh - 1;
    if (hi < 0 || hi >= 112) continue;
    for (int kww = 0; kww < 3; ++kww) {
      const int wi = wo * 2 + kww - 1;
      if (wi < 0 || wi >= 112) continue;
      best = fmaxf(best, bf2f(in[((size_t)(b * 112 + hi) * 112 + wi) * 64 + c]));
    }
  }
  out[t] = f2bf(best);
}

// ============================================================================
// LSTM packing + recurrence
// ============================================================================
__global__ void k_pack_wih(const float* __restrict__ wih, bf16_t* __restrict__ wp) {
  const long t = (long)blockIdx.x * 256 + threadIdx.x;
  if (t >= 14336L * 1024) return;
  const long k = t / 1024; const int n = (int)(t % 1024);
  wp[t] = f2bf(wih[(long)n * 14336 + k]);                 // B[k][n] = wih[n][k]
}
__global__ void k_pack_whh(const float* __restrict__ whh, bf16_t* __restrict__ wp) {
  const int t = blockIdx.x * 256 + threadIdx.x;
  if (t >= 256 * 1024) return;
  const int k = t / 1024, n = t % 1024;
  wp[t] = f2bf(whh[n * 256 + k]);
}
__global__ void k_bias2(const float* __restrict__ a, const float* __restrict__ b,
                        float* __restrict__ o) {
  const int t = blockIdx.x * 256 + threadIdx.x;
  if (t < 1024) o[t] = a[t] + b[t];
}
// A[(b*7+r)][d], d=c*7+w  <-  feat NHWC [64][7][7][2048] bf16
__global__ void k_pack_A(const bf16_t* __restrict__ feat, bf16_t* __restrict__ A) {
  const long t = (long)blockIdx.x * 256 + threadIdx.x;
  if (t >= 448L * 14336) return;
  const long m = t / 14336; const int d = (int)(t % 14336);
  const int b = (int)(m / 7), r = (int)(m % 7);
  const int c = d / 7, w = d % 7;
  A[t] = feat[(((size_t)(b * 7 + r)) * 7 + w) * 2048 + c];
}

// 7 timesteps, one workgroup (8 waves). gates = pre[:,t,:] + h @ whh^T (WMMA).
__global__ __launch_bounds__(256) void k_lstm_steps(const float* __restrict__ pre,
    const bf16_t* __restrict__ whhp, bf16_t* __restrict__ hG, float* __restrict__ cG,
    float* __restrict__ gatesG, float* __restrict__ lo) {
  const int tid = threadIdx.x;
  for (int e = tid; e < 64 * 256; e += 256) { hG[e] = 0; cG[e] = 0.f; }
  __threadfence(); __syncthreads();
  const int wave = tid >> 5, lane = tid & 31, l15 = lane & 15, lhi = lane >> 4;
  for (int t = 0; t < 7; ++t) {
    for (int job = wave; job < 256; job += 8) {
      const int mt = (job & 3) * 16;        // batch tile
      const int nt = (job >> 2) * 16;       // gate-column tile
      const int n  = nt + l15;
      const int mb = mt + lhi * 8;
      v8f acc;
      for (int r = 0; r < 8; ++r) acc[r] = pre[((size_t)(mb + r) * 7 + t) * 1024 + n];
      for (int k0 = 0; k0 < 256; k0 += 32) {
        const unsigned char* ap = (const unsigned char*)hG +
            (size_t)(mt + l15) * 512 + (size_t)(k0 + lhi * 8) * 2;
        v8i a = mk8(*(const int4*)ap, *(const int4*)(ap + 32));
        const unsigned char* bp = (const unsigned char*)whhp +
            ((size_t)(k0 + lhi * 8) * 1024 + nt + l15) * 2;
        v8i b;
        for (int j = 0; j < 8; ++j) {
          const unsigned char* q0 = bp + (size_t)((j & 3) * 2 + (j >> 2) * 16) * 2048;
          b[j] = (int)((unsigned)*(const unsigned short*)q0 | ((unsigned)*(const unsigned short*)(q0 + 2048) << 16));
        }
        acc = wmma_bf16(a, b, acc);
      }
      for (int r = 0; r < 8; ++r) gatesG[(size_t)(mb + r) * 1024 + n] = acc[r];
    }
    __threadfence(); __syncthreads();
    for (int e = tid; e < 64 * 256; e += 256) {
      const int b = e >> 8, j = e & 255;
      const float gi = gatesG[b * 1024 + j],       gf = gatesG[b * 1024 + 256 + j];
      const float gg = gatesG[b * 1024 + 512 + j], go = gatesG[b * 1024 + 768 + j];
      const float c = sigf(gf) * cG[e] + sigf(gi) * tanhf(gg);
      const float h = sigf(go) * tanhf(c);
      cG[e] = c; hG[e] = f2bf(h);
      lo[((size_t)b * 7 + t) * 256 + j] = h;
    }
    __threadfence(); __syncthreads();
  }
}

// attention + pooled tanh + final linear -> [64][23] f32
__global__ __launch_bounds__(256) void k_head(const float* __restrict__ lo,
    const float* __restrict__ attw, const float* __restrict__ attb,
    const float* __restrict__ mw, const float* __restrict__ mb, float* __restrict__ out) {
  const int b = blockIdx.x, tid = threadIdx.x;
  __shared__ float sc[8], at[8], pooled[256];
  const int wv = tid >> 5, ln = tid & 31;
  if (wv < 7) {
    float s = 0.f;
    for (int h = ln; h < 256; h += 32)
      s += attw[h] * tanhf(lo[((size_t)b * 7 + wv) * 256 + h]);
    for (int m = 16; m > 0; m >>= 1) s += __shfl_xor(s, m, 32);
    if (ln == 0) sc[wv] = s + attb[0];
  }
  __syncthreads();
  if (tid == 0) {
    float mx = sc[0];
    for (int t = 1; t < 7; ++t) mx = fmaxf(mx, sc[t]);
    float e[7], sum = 0.f;
    for (int t = 0; t < 7; ++t) { e[t] = expf(sc[t] - mx); sum += e[t]; }
    for (int t = 0; t < 7; ++t) at[t] = e[t] / sum;
  }
  __syncthreads();
  {
    float pv = 0.f;
    for (int t = 0; t < 7; ++t) pv += at[t] * lo[((size_t)b * 7 + t) * 256 + tid];
    pooled[tid] = tanhf(pv);
  }
  __syncthreads();
  if (tid < 23) {
    float o = mb[tid];
    for (int h = 0; h < 256; ++h) o += mw[tid * 256 + h] * pooled[h];
    out[b * 23 + tid] = o;
  }
}

// ============================================================================
// Host orchestration
// ============================================================================
extern "C" void kernel_launch(void* const* d_in, const int* in_sizes, int n_in,
                              void* d_out, int out_size, void* d_ws, size_t ws_size,
                              hipStream_t stream) {
  (void)in_sizes; (void)n_in; (void)out_size; (void)ws_size;
  // -------- leaf map: jax tree_flatten (sorted dict keys), 274 leaves --------
  // 0:data  1:att_b 2:att_w  3..6: stem bn {b,g,m,v}  7:conv1  8..267:layers
  // 268:lstm.bhh 269:lstm.bih 270:lstm.whh 271:lstm.wih  272:mlp2_b 273:mlp2_w
  auto L = [&](int i) -> const float* { return (const float*)d_in[i]; };
  const float* data   = L(0);
  const float* att_b  = L(1);
  const float* att_w  = L(2);
  const float* sb = L(3), *sg = L(4), *smn = L(5), *sv = L(6);
  const float* conv1w = L(7);
  const float* lstm_bhh = L(268), *lstm_bih = L(269);
  const float* lstm_whh = L(270), *lstm_wih = L(271);
  const float* mlp_b = L(272), *mlp_w = L(273);

  // -------- workspace bump allocator --------
  size_t off = 0;
  auto alloc = [&](size_t bytes) -> void* {
    void* p = (void*)((char*)d_ws + off);
    off += (bytes + 255) & ~(size_t)255;
    return p;
  };
  const size_t BIGE = 64UL * 56 * 56 * 256;                 // 51,380,224 elems
  bf16_t* B0 = (bf16_t*)alloc(BIGE * 2);
  bf16_t* B1 = (bf16_t*)alloc(BIGE * 2);
  bf16_t* BR = (bf16_t*)alloc(BIGE * 2);
  bf16_t* T1 = (bf16_t*)alloc(64UL * 56 * 56 * 128 * 2);
  bf16_t* T2 = (bf16_t*)alloc(64UL * 56 * 56 * 64 * 2);
  float* w1f   = (float*)alloc(64 * 147 * 4);
  float* bias1 = (float*)alloc(64 * 4);

  // -------- stem --------
  k_fold_conv1<<<(64 * 147 + 255) / 256, 256, 0, stream>>>(conv1w, sg, sb, smn, sv, w1f, bias1);
  k_conv1<<<(unsigned)((802816UL * 64) / 256), 256, 0, stream>>>(data, w1f, bias1, B1);
  k_pool<<<(unsigned)((64UL * 56 * 56 * 64) / 256), 256, 0, stream>>>(B1, B0);

  // -------- generic conv runner (prep + WMMA gemm) --------
  auto run_conv = [&](const bf16_t* in, int Hin, int Win, int Cin, int Cout,
                      int KH, int KW, int stride, int pad,
                      const float* w, const float* g, const float* bb,
                      const float* bm, const float* bv,
                      const bf16_t* res, bf16_t* out_bf, float* out_f, int relu) {
    const int Ho = (Hin + 2 * pad - KH) / stride + 1;
    const int Wo = (Win + 2 * pad - KW) / stride + 1;
    const long wtot = (long)Cout * Cin * KH * KW;
    bf16_t* wp  = (bf16_t*)alloc((size_t)wtot * 2);
    float* bias = (float*)alloc((size_t)Cout * 4);
    k_fold_conv<<<(unsigned)((wtot + 255) / 256), 256, 0, stream>>>(
        w, g, bb, bm, bv, wp, bias, Cout, Cin, KH, KW);
    ConvP q{};
    q.in = in; q.wp = wp; q.bias = bias; q.res = res; q.out_bf = out_bf; q.out_f = out_f;
    q.Hin = Hin; q.Win = Win; q.Cin = Cin; q.Ho = Ho; q.Wo = Wo; q.Cout = Cout;
    q.KH = KH; q.KW = KW; q.stride = stride; q.pad = pad;
    q.M = 64 * Ho * Wo; q.steps = KH * KW * Cin / 32; q.relu = relu;
    q.out_f32 = (out_f != nullptr);
    dim3 grid(Cout / 64, (q.M + TM - 1) / TM);
    k_conv_wmma<<<grid, 256, 0, stream>>>(q);
  };

  // -------- bottleneck stages --------
  const int planesA[4] = {64, 128, 256, 512};
  const int nA[4]      = {3, 4, 6, 3};
  const int sA[4]      = {1, 2, 2, 2};
  bf16_t* X = B0; bf16_t* Y = B1;
  int Cin = 64, H = 56, W = 56, leaf = 8;
  for (int si = 0; si < 4; ++si) {
    for (int bi = 0; bi < nA[si]; ++bi) {
      const int  s  = (bi == 0) ? sA[si] : 1;
      const bool ds = (bi == 0);
      const int  P = planesA[si], Cout = 4 * P;
      const int  Ho = (H + 2 - 3) / s + 1;       // spatial after the 3x3 (pad 1)
      // leaves within block: bn1{b,g,m,v} bn2{..} bn3{..} [bnd{..}] w1 w2 w3 [wd]
      const int base = leaf;
      const float *b1b=L(base+0),  *b1g=L(base+1),  *b1m=L(base+2),  *b1v=L(base+3);
      const float *b2b=L(base+4),  *b2g=L(base+5),  *b2m=L(base+6),  *b2v=L(base+7);
      const float *b3b=L(base+8),  *b3g=L(base+9),  *b3m=L(base+10), *b3v=L(base+11);
      const int wbase = base + (ds ? 16 : 12);
      const float *w1 = L(wbase), *w2 = L(wbase + 1), *w3 = L(wbase + 2);

      run_conv(X, H, W, Cin, P, 1, 1, 1, 0, w1, b1g, b1b, b1m, b1v, nullptr, T1, nullptr, 1);
      run_conv(T1, H, W, P, P, 3, 3, s, 1, w2, b2g, b2b, b2m, b2v, nullptr, T2, nullptr, 1);
      const bf16_t* res;
      if (ds) {
        const float *bdb=L(base+12), *bdg=L(base+13), *bdm=L(base+14), *bdv=L(base+15);
        const float *wd = L(wbase + 3);
        run_conv(X, H, W, Cin, Cout, 1, 1, s, 0, wd, bdg, bdb, bdm, bdv, nullptr, BR, nullptr, 0);
        res = BR;
      } else res = X;
      run_conv(T2, Ho, Ho, P, Cout, 1, 1, 1, 0, w3, b3g, b3b, b3m, b3v, res, Y, nullptr, 1);

      bf16_t* t = X; X = Y; Y = t;
      Cin = Cout; H = Ho; W = Ho;
      leaf += ds ? 20 : 15;
    }
  }
  // X now holds features [64,7,7,2048] NHWC bf16

  // -------- LSTM --------
  bf16_t* wihp  = (bf16_t*)alloc(14336UL * 1024 * 2);
  bf16_t* whhp  = (bf16_t*)alloc(256UL * 1024 * 2);
  float*  lbias = (float*)alloc(1024 * 4);
  bf16_t* Albuf = (bf16_t*)alloc(448UL * 14336 * 2);
  float*  pre   = (float*)alloc(448UL * 1024 * 4);
  bf16_t* hG    = (bf16_t*)alloc(64 * 256 * 2);
  float*  cG    = (float*)alloc(64 * 256 * 4);
  float*  gatesG= (float*)alloc(64 * 1024 * 4);
  float*  lo    = (float*)alloc(64UL * 7 * 256 * 4);

  k_pack_wih<<<(unsigned)((14336UL * 1024) / 256), 256, 0, stream>>>(lstm_wih, wihp);
  k_pack_whh<<<(256 * 1024) / 256, 256, 0, stream>>>(lstm_whh, whhp);
  k_bias2<<<4, 256, 0, stream>>>(lstm_bih, lstm_bhh, lbias);
  k_pack_A<<<(unsigned)((448UL * 14336) / 256), 256, 0, stream>>>(X, Albuf);

  // pre-gates = A @ wih^T + (bih+bhh): M=448, N=1024, K=14336 through the WMMA kernel
  ConvP q{};
  q.in = Albuf; q.wp = wihp; q.bias = lbias; q.res = nullptr;
  q.out_bf = nullptr; q.out_f = pre;
  q.Hin = 1; q.Win = 1; q.Cin = 14336; q.Ho = 1; q.Wo = 1; q.Cout = 1024;
  q.KH = 1; q.KW = 1; q.stride = 1; q.pad = 0;
  q.M = 448; q.steps = 14336 / 32; q.relu = 0; q.out_f32 = 1;
  {
    dim3 grid(1024 / 64, (448 + TM - 1) / TM);
    k_conv_wmma<<<grid, 256, 0, stream>>>(q);
  }
  k_lstm_steps<<<1, 256, 0, stream>>>(pre, whhp, hG, cG, gatesG, lo);

  // -------- attention + MLP head --------
  k_head<<<64, 256, 0, stream>>>(lo, att_w, att_b, mlp_w, mlp_b, (float*)d_out);
}
